// LoRAXFormersSpatialTemporalAttnProcessor_65420941853295
// MI455X (gfx1250) — matrix operand order repaired and checked by reading, not compiled, not run on hardware
//
#include <hip/hip_runtime.h>
#include <hip/hip_bf16.h>
#include <stdint.h>

#define D_MODEL 1280
#define M_TOK   4096
#define N_HEADS 8
#define D_HEAD  160
#define LORA_R  4
#define LORA_SCALE 1.0f

typedef __attribute__((ext_vector_type(16))) __bf16       v16bf;
typedef __attribute__((ext_vector_type(8)))  float        v8f;
typedef __attribute__((ext_vector_type(4)))  unsigned int u32x4;
typedef __attribute__((ext_vector_type(2)))  unsigned int u32x2;
typedef __attribute__((ext_vector_type(4)))  float        f32x4;
typedef __attribute__((ext_vector_type(8)))  int          i32x8;
typedef __attribute__((ext_vector_type(4)))  int          i32x4;

union AF { v16bf v; u32x4 q[2]; unsigned short h[16]; };

__device__ __forceinline__ unsigned short f2bf(float f) {
  union { float f; unsigned int u; } c; c.f = f;
  unsigned int u = c.u;
  u += 0x7FFFu + ((u >> 16) & 1u);   // round-to-nearest-even
  return (unsigned short)(u >> 16);
}

#if defined(__has_builtin)
#if __has_builtin(__builtin_amdgcn_tensor_load_to_lds)
#define HAVE_TDM 1
#endif
#endif

#ifdef HAVE_TDM
// 2D strided tile (elements are 2 bytes) -> LDS, per CDNA5 D# layout (ISA ch.10.8/§8).
// group0: [1:0]=count=1, [63:32]=lds byte addr, [120:64]=global addr, [127:126]=type 2.
// group1: [17:16]=data_size(1->2B), dim0/dim1, tile0/tile1, dim0 stride. groups 2/3: zero (2D).
__device__ __forceinline__ void tdm_load_2d_bf16(unsigned lds_off, const void* gptr,
                                                 unsigned tile0, unsigned tile1,
                                                 unsigned tdim0, unsigned tdim1,
                                                 unsigned stride0)
{
  unsigned long long ga = (unsigned long long)(uintptr_t)gptr;
  u32x4 g0;
  g0.x = 1u;                                        // count=1, user descriptor
  g0.y = lds_off;                                   // LDS byte address
  g0.z = (unsigned)ga;                              // global addr [31:0]
  g0.w = (unsigned)((ga >> 32) & 0x1FFFFFFu) | (2u << 30);  // [56:32] + type=2
  i32x8 g1;
  g1[0] = (int)(1u << 16);                          // data_size = 1 (2 bytes)
  g1[1] = (int)((tdim0 & 0xFFFFu) << 16);           // tensor_dim0[15:0]
  g1[2] = (int)((tdim0 >> 16) | ((tdim1 & 0xFFFFu) << 16));
  g1[3] = (int)((tdim1 >> 16) | (tile0 << 16));     // tile_dim0
  g1[4] = (int)tile1;                               // tile_dim1 (tile_dim2 = 0)
  g1[5] = (int)stride0;                             // tensor_dim0_stride[31:0]
  g1[6] = 0;
  g1[7] = 0;
  i32x4 z4 = {0, 0, 0, 0};
#if defined(__clang_major__) && __clang_major__ >= 23
  i32x8 z8 = {0, 0, 0, 0, 0, 0, 0, 0};
  __builtin_amdgcn_tensor_load_to_lds(g0, g1, z4, z4, z8, 0);
#else
  __builtin_amdgcn_tensor_load_to_lds(g0, g1, z4, z4, 0);
#endif
}
#endif

// ---------------------------------------------------------------------------
// Generic bf16 WMMA GEMM: C[M,N] = (A[M,K] @ W[K,N] + LORA_SCALE*t[M,4]@Bup[4,N])*scale + bias
// A is f32 or bf16 (template), W/bias/t/Bup f32, C is bf16 or f32 (template).
// Block tile 128x128, BK=32, 256 threads = 8 waves, each wave 32x64 (2x4 WMMAs).
// ---------------------------------------------------------------------------
template<bool A_BF16, bool OUT_BF16>
__global__ __launch_bounds__(256)
void gemm_wmma(const void* __restrict__ Ag, const float* __restrict__ Wg,
               const float* __restrict__ bias,
               const float* __restrict__ tdown, const float* __restrict__ Bup,
               void* __restrict__ Cg, float scale, int Mdim, int Ndim, int Kdim)
{
  __shared__ unsigned short lA[128 * 40];   // [m][k], row stride 40 (pad)
  __shared__ unsigned short lB[128 * 40];   // [n][k] (W transposed), stride 40

  const int tid  = threadIdx.x;
  const int lane = tid & 31, wid = tid >> 5;
  const int g = lane >> 4, l16 = lane & 15;
  const int wm = wid >> 1, wn = wid & 1;
  const int blockN = blockIdx.x * 128, blockM = blockIdx.y * 128;

  v8f c[2][4];
  for (int mi = 0; mi < 2; ++mi)
    for (int ni = 0; ni < 4; ++ni)
      for (int e = 0; e < 8; ++e) c[mi][ni][e] = 0.f;

  const int ar0 = tid >> 3, ac4 = (tid & 7) * 4;     // A loader: 128 rows x 32 k
  const int bn4 = (tid & 31) * 4, bk0 = tid >> 5;    // B loader: 32 k x 128 n

  for (int k0 = 0; k0 < Kdim; k0 += 32) {
    __syncthreads();
    // ---- stage A tile (convert f32->bf16 if needed) ----
    for (int i = 0; i < 4; ++i) {
      int r = ar0 + 32 * i;
      if constexpr (A_BF16) {
        u32x2 raw = *(const u32x2*)((const unsigned short*)Ag +
                                    (size_t)(blockM + r) * Kdim + k0 + ac4);
        *(u32x2*)&lA[r * 40 + ac4] = raw;
      } else {
        f32x4 v = *(const f32x4*)((const float*)Ag +
                                  (size_t)(blockM + r) * Kdim + k0 + ac4);
        u32x2 p;
        p.x = (unsigned)f2bf(v.x) | ((unsigned)f2bf(v.y) << 16);
        p.y = (unsigned)f2bf(v.z) | ((unsigned)f2bf(v.w) << 16);
        *(u32x2*)&lA[r * 40 + ac4] = p;
      }
    }
    // ---- stage W tile transposed: lB[n][k] ----
    for (int i = 0; i < 4; ++i) {
      int k = bk0 + 8 * i;
      f32x4 v = *(const f32x4*)(Wg + (size_t)(k0 + k) * Ndim + blockN + bn4);
      lB[(bn4 + 0) * 40 + k] = f2bf(v.x);
      lB[(bn4 + 1) * 40 + k] = f2bf(v.y);
      lB[(bn4 + 2) * 40 + k] = f2bf(v.z);
      lB[(bn4 + 3) * 40 + k] = f2bf(v.w);
    }
    if (k0 + 32 < Kdim)   // gfx1250 global_prefetch_b8 on the weight stream
      __builtin_prefetch(Wg + (size_t)(k0 + 32 + bk0) * Ndim + blockN + bn4, 0, 1);
    __syncthreads();

    // ---- fragments: A and B lane-layout (K chunks at 8g / 16+8g) ----
    AF a[2], b[4];
    for (int mi = 0; mi < 2; ++mi) {
      int row = wm * 32 + mi * 16 + l16;
      a[mi].q[0] = *(const u32x4*)&lA[row * 40 + 8 * g];
      a[mi].q[1] = *(const u32x4*)&lA[row * 40 + 16 + 8 * g];
    }
    for (int ni = 0; ni < 4; ++ni) {
      int n = wn * 64 + ni * 16 + l16;
      b[ni].q[0] = *(const u32x4*)&lB[n * 40 + 8 * g];
      b[ni].q[1] = *(const u32x4*)&lB[n * 40 + 16 + 8 * g];
    }
    for (int mi = 0; mi < 2; ++mi)
      for (int ni = 0; ni < 4; ++ni)
        c[mi][ni] = __builtin_amdgcn_wmma_f32_16x16x32_bf16(
            false, a[mi].v, false, b[ni].v, (short)0, c[mi][ni], false, false);
  }

  // ---- epilogue: LoRA-up + scale + bias, store ----
  for (int mi = 0; mi < 2; ++mi)
    for (int ni = 0; ni < 4; ++ni) {
      int gcol = blockN + wn * 64 + ni * 16 + l16;
      for (int e = 0; e < 8; ++e) {
        int grow = blockM + wm * 32 + mi * 16 + e + 8 * g;
        float val = c[mi][ni][e];
        if (tdown) {
          float l = 0.f;
          for (int r = 0; r < LORA_R; ++r)
            l += tdown[(size_t)grow * LORA_R + r] * Bup[(size_t)r * Ndim + gcol];
          val += LORA_SCALE * l;
        }
        val *= scale;
        if (bias) val += bias[gcol];
        if constexpr (OUT_BF16)
          ((unsigned short*)Cg)[(size_t)grow * Ndim + gcol] = f2bf(val);
        else
          ((float*)Cg)[(size_t)grow * Ndim + gcol] = val;
      }
    }
}

// ---------------------------------------------------------------------------
// Flash-style attention: block = (128 query rows, 1 head), 8 waves x 16 rows.
// K tile staged via TDM (tensor_load_to_lds), V staged manually transposed.
// Per 32-key step: 10 score WMMAs (dh=160), online softmax via LDS, 10 PV WMMAs.
// Q pre-scaled by dh^-0.5 at projection time. Q/K/V/O are bf16, M x 1280.
// ---------------------------------------------------------------------------
__global__ __launch_bounds__(256)
void attn_wmma(const unsigned short* __restrict__ Q, const unsigned short* __restrict__ K,
               const unsigned short* __restrict__ V, unsigned short* __restrict__ O)
{
  __shared__ unsigned short lK [32 * 160];      // [token][dh], natural (TDM dest)
  __shared__ unsigned short lVt[160 * 40];      // [dh][token] (transposed)
  __shared__ float          lS [8][16 * 33];    // per-wave raw scores
  __shared__ unsigned short lP [8][16 * 40];    // per-wave exp(P) bf16
  __shared__ float          lFac[8][16];        // per-row rescale factors

  const int tid = threadIdx.x, lane = tid & 31, wid = tid >> 5;
  const int g = lane >> 4, l16 = lane & 15;
  const int head = blockIdx.y;
  const int q0 = blockIdx.x * 128;
  const int qrow = q0 + wid * 16 + l16;

  AF aq[5];                                      // Q fragments, dh = 5 x 32
  for (int ks = 0; ks < 5; ++ks) {
    size_t base = (size_t)qrow * D_MODEL + head * D_HEAD + ks * 32;
    aq[ks].q[0] = *(const u32x4*)(Q + base + 8 * g);
    aq[ks].q[1] = *(const u32x4*)(Q + base + 16 + 8 * g);
  }

  v8f o[10];
  for (int ni = 0; ni < 10; ++ni) for (int e = 0; e < 8; ++e) o[ni][e] = 0.f;

  float rmax = -1e30f, rsum = 0.f;
  const int sw = tid >> 4, srow = tid & 15;      // softmax ownership (tid < 128)

#ifdef HAVE_TDM
  const unsigned lK_off = (unsigned)(uintptr_t)&lK[0];   // LDS byte offset (addr[31:0])
#endif

  for (int kb = 0; kb < M_TOK / 32; ++kb) {
    __syncthreads();                             // prior reads of lK/lVt complete
    // ---- K tile: 32 tokens x 160 dh, row stride D_MODEL, via Tensor Data Mover ----
#ifdef HAVE_TDM
    if (wid == 0)
      tdm_load_2d_bf16(lK_off,
                       K + (size_t)(kb * 32) * D_MODEL + head * D_HEAD,
                       /*tile0=*/D_HEAD, /*tile1=*/32,
                       /*tdim0=*/D_MODEL, /*tdim1=*/M_TOK, /*stride0=*/D_MODEL);
#endif
    // ---- V tile: manual transposed staging (TDM has no transpose) ----
    for (int i = 0; i < 5; ++i) {
      int v4 = tid + 256 * i;                    // 0..1279 quads
      int tok = v4 / 40, c4 = (v4 % 40) * 4;
      size_t gidx = (size_t)(kb * 32 + tok) * D_MODEL + head * D_HEAD + c4;
#ifndef HAVE_TDM
      u32x2 kraw = *(const u32x2*)(K + gidx);
      *(u32x2*)&lK[tok * 160 + c4] = kraw;
#endif
      u32x2 vraw = *(const u32x2*)(V + gidx);
      lVt[(c4 + 0) * 40 + tok] = (unsigned short)(vraw.x & 0xFFFFu);
      lVt[(c4 + 1) * 40 + tok] = (unsigned short)(vraw.x >> 16);
      lVt[(c4 + 2) * 40 + tok] = (unsigned short)(vraw.y & 0xFFFFu);
      lVt[(c4 + 3) * 40 + tok] = (unsigned short)(vraw.y >> 16);
    }
#ifdef HAVE_TDM
    if (wid == 0)
      __builtin_amdgcn_s_wait_tensorcnt(0);      // TDM done before barrier publishes
#endif
    __syncthreads();

    // ---- scores: S[16q x 32k] as two 16x16 C-tiles ----
    v8f s[2];
    for (int sub = 0; sub < 2; ++sub) for (int e = 0; e < 8; ++e) s[sub][e] = 0.f;
    for (int ks = 0; ks < 5; ++ks)
      for (int sub = 0; sub < 2; ++sub) {
        AF bk;
        int tok = sub * 16 + l16;
        bk.q[0] = *(const u32x4*)&lK[tok * 160 + ks * 32 + 8 * g];
        bk.q[1] = *(const u32x4*)&lK[tok * 160 + ks * 32 + 16 + 8 * g];
        s[sub] = __builtin_amdgcn_wmma_f32_16x16x32_bf16(
            false, aq[ks].v, false, bk.v, (short)0, s[sub], false, false);
      }
    for (int sub = 0; sub < 2; ++sub)
      for (int e = 0; e < 8; ++e)
        lS[wid][(e + 8 * g) * 33 + sub * 16 + l16] = s[sub][e];
    __syncthreads();

    // ---- online softmax: one thread per query row ----
    if (tid < 128) {
      float sc[32], mx = -1e30f;
      for (int j = 0; j < 32; ++j) { sc[j] = lS[sw][srow * 33 + j]; mx = fmaxf(mx, sc[j]); }
      float mnew = fmaxf(rmax, mx);
      float r = __expf(rmax - mnew);
      float ps = 0.f;
      for (int j = 0; j < 32; ++j) {
        float p = __expf(sc[j] - mnew);
        ps += p;
        lP[sw][srow * 40 + j] = f2bf(p);
      }
      rsum = rsum * r + ps;
      rmax = mnew;
      lFac[sw][srow] = r;
    }
    __syncthreads();

    // ---- rescale O, then O += P @ V ----
    float rs[8];
    for (int e = 0; e < 8; ++e) rs[e] = lFac[wid][e + 8 * g];
    for (int ni = 0; ni < 10; ++ni)
      for (int e = 0; e < 8; ++e) o[ni][e] *= rs[e];
    AF ap;
    ap.q[0] = *(const u32x4*)&lP[wid][l16 * 40 + 8 * g];
    ap.q[1] = *(const u32x4*)&lP[wid][l16 * 40 + 16 + 8 * g];
    for (int ni = 0; ni < 10; ++ni) {
      AF bv;
      int dcol = ni * 16 + l16;
      bv.q[0] = *(const u32x4*)&lVt[dcol * 40 + 8 * g];
      bv.q[1] = *(const u32x4*)&lVt[dcol * 40 + 16 + 8 * g];
      o[ni] = __builtin_amdgcn_wmma_f32_16x16x32_bf16(
          false, ap.v, false, bv.v, (short)0, o[ni], false, false);
    }
  }

  // ---- final 1/sum normalization + store bf16 ----
  __syncthreads();
  if (tid < 128) lFac[sw][srow] = 1.f / rsum;
  __syncthreads();
  float rn[8];
  for (int e = 0; e < 8; ++e) rn[e] = lFac[wid][e + 8 * g];
  for (int ni = 0; ni < 10; ++ni) {
    int dcol = head * D_HEAD + ni * 16 + l16;
    for (int e = 0; e < 8; ++e) {
      int row = q0 + wid * 16 + e + 8 * g;
      O[(size_t)row * D_MODEL + dcol] = f2bf(o[ni][e] * rn[e]);
    }
  }
}

// ---------------------------------------------------------------------------
// Rank-4 LoRA down-projection: T[M,4] = X[M,K] @ A[K,4]
// ---------------------------------------------------------------------------
__global__ __launch_bounds__(256)
void lora_down(const float* __restrict__ X, const float* __restrict__ A,
               float* __restrict__ T, int Kdim)
{
  __shared__ float red[4][256];
  int row = blockIdx.x, tid = threadIdx.x;
  float a0 = 0, a1 = 0, a2 = 0, a3 = 0;
  for (int i = tid; i < Kdim; i += 256) {
    float x = X[(size_t)row * Kdim + i];
    f32x4 ar = *(const f32x4*)(A + i * 4);
    a0 += x * ar.x; a1 += x * ar.y; a2 += x * ar.z; a3 += x * ar.w;
  }
  red[0][tid] = a0; red[1][tid] = a1; red[2][tid] = a2; red[3][tid] = a3;
  __syncthreads();
  for (int s = 128; s > 0; s >>= 1) {
    if (tid < s)
      for (int r = 0; r < 4; ++r) red[r][tid] += red[r][tid + s];
    __syncthreads();
  }
  if (tid < 4) T[row * 4 + tid] = red[tid][0];
}

// Out[M,N] += LORA_SCALE * T[M,4] @ Bup[4,N]
__global__ __launch_bounds__(256)
void lora_up_add(float* __restrict__ Out, const float* __restrict__ T,
                 const float* __restrict__ Bup)
{
  int idx = blockIdx.x * 256 + threadIdx.x;
  int row = idx / D_MODEL, col = idx - row * D_MODEL;
  f32x4 t = *(const f32x4*)(T + row * 4);
  float l = t.x * Bup[col] + t.y * Bup[D_MODEL + col] +
            t.z * Bup[2 * D_MODEL + col] + t.w * Bup[3 * D_MODEL + col];
  Out[idx] += LORA_SCALE * l;
}

// ---------------------------------------------------------------------------
extern "C" void kernel_launch(void* const* d_in, const int* in_sizes, int n_in,
                              void* d_out, int out_size, void* d_ws, size_t ws_size,
                              hipStream_t stream)
{
  (void)in_sizes; (void)n_in; (void)out_size; (void)ws_size;
  const float* x  = (const float*)d_in[0];
  const float* Wq = (const float*)d_in[1];
  const float* Wk = (const float*)d_in[2];
  const float* Wv = (const float*)d_in[3];
  const float* Wo = (const float*)d_in[4];
  const float* bo = (const float*)d_in[5];
  const float* Aq = (const float*)d_in[6];
  const float* Bq = (const float*)d_in[7];
  const float* Ak = (const float*)d_in[8];
  const float* Bk = (const float*)d_in[9];
  const float* Av = (const float*)d_in[10];
  const float* Bv = (const float*)d_in[11];
  const float* Ao = (const float*)d_in[12];
  const float* Bo = (const float*)d_in[13];

  const size_t MD = (size_t)M_TOK * D_MODEL;
  unsigned short* Qb = (unsigned short*)d_ws;   // bf16 Q (pre-scaled)
  unsigned short* Kb = Qb + MD;
  unsigned short* Vb = Kb + MD;
  unsigned short* Ob = Vb + MD;                 // bf16 attention output
  float* tq = (float*)(Ob + MD);
  float* tk = tq + M_TOK * LORA_R;
  float* tv = tk + M_TOK * LORA_R;
  float* to = tv + M_TOK * LORA_R;

  const float qscale = 0.07905694150420949f;    // 160^-0.5 folded into Q

  lora_down<<<M_TOK, 256, 0, stream>>>(x, Aq, tq, D_MODEL);
  lora_down<<<M_TOK, 256, 0, stream>>>(x, Ak, tk, D_MODEL);
  lora_down<<<M_TOK, 256, 0, stream>>>(x, Av, tv, D_MODEL);

  dim3 gg(D_MODEL / 128, M_TOK / 128);
  gemm_wmma<false, true><<<gg, 256, 0, stream>>>(x, Wq, nullptr, tq, Bq, Qb, qscale, M_TOK, D_MODEL, D_MODEL);
  gemm_wmma<false, true><<<gg, 256, 0, stream>>>(x, Wk, nullptr, tk, Bk, Kb, 1.f,    M_TOK, D_MODEL, D_MODEL);
  gemm_wmma<false, true><<<gg, 256, 0, stream>>>(x, Wv, nullptr, tv, Bv, Vb, 1.f,    M_TOK, D_MODEL, D_MODEL);

  attn_wmma<<<dim3(M_TOK / 128, N_HEADS), 256, 0, stream>>>(Qb, Kb, Vb, Ob);

  gemm_wmma<true, false><<<gg, 256, 0, stream>>>(Ob, Wo, bo, nullptr, nullptr, d_out, 1.f, M_TOK, D_MODEL, D_MODEL);

  lora_down<<<M_TOK, 256, 0, stream>>>((const float*)d_out, Ao, to, D_MODEL);
  lora_up_add<<<MD / 256, 256, 0, stream>>>((float*)d_out, to, Bo);
}
